// ModelGnnPinn_74586402062378
// MI455X (gfx1250) — compile-verified, matrix operand order checked by reading, not run on hardware
//
#include <hip/hip_runtime.h>
#include <hip/hip_bf16.h>
#include <stdint.h>

typedef __attribute__((ext_vector_type(16))) _Float16 v16h;
typedef __attribute__((ext_vector_type(8)))  _Float16 v8h;
typedef __attribute__((ext_vector_type(4)))  _Float16 v4h;
typedef __attribute__((ext_vector_type(8)))  float    v8f;
typedef __attribute__((ext_vector_type(4)))  float    v4f;

#define N_NODES 32000
#define N_EDGES 512000
#define LAT 32
#define HID 128
#define MP_ITERS 4
#define MT 64           // rows per workgroup tile (32000/64=500, 512000/64=8000 exact)
#define XSTR_ENC 40     // padded LDS row strides (halfs); all 16B-aligned, bank-friendly
#define XSTR_EDGE 104
#define XSTR_NODE 72
#define HSTR 136

// ---------- WMMA fragment helper (gfx1250 wave32 layouts) ----------
// k-contiguous fragment: lane l -> row l%16 of the (16 x K) operand, h=l/16;
// element e covers k = (e<8 ? h*8+e : 16+h*8+(e-8)). Works for A rows (W^T[n][k])
// and B columns (row-major X[m][k]) identically.
__device__ inline v16h load_frag(const _Float16* row, int h, int k0) {
  v8h lo = *(const v8h*)(row + k0 + h * 8);
  v8h hi = *(const v8h*)(row + k0 + 16 + h * 8);
  return __builtin_shufflevector(lo, hi, 0,1,2,3,4,5,6,7,8,9,10,11,12,13,14,15);
}

// Transposed-product tile: D = W_tile^T-frag x X_tile-frag  (i.e. O^T tile).
// Resulting lane mapping: l%16 = output ROW m, element e -> output col n = 8*(l/16)+e.
template<int KP>
__device__ inline v8f tile_mm_t(const _Float16* xrow, const _Float16* wrow, int h) {
  v8f acc = {};
  #pragma unroll
  for (int k0 = 0; k0 < KP; k0 += 32) {
    v16h w = load_frag(wrow, h, k0);
    v16h x = load_frag(xrow, h, k0);
    acc = __builtin_amdgcn_wmma_f32_16x16x32_f16(false, w, false, x, (short)0, acc,
                                                 false, false);
  }
  return acc;
}

// One hidden layer: O[MT x 128] = relu(X[MT x KP] @ W[KP x 128]), f16 in LDS.
// Computed as O^T = W^T @ X^T. Wave w owns row-subtile tm = w&3 (X fragments
// loaded ONCE into VGPRs) and sweeps 4 column tiles tn = (w>>2)*4..+3.
// Epilogue: convert to f16 (v_cvt_pk), relu in f16 domain (v_pk_max, no
// canonicalize needed on cvt results), single ds_store_b128 per lane per tile.
template<int KP>
__device__ inline void mlp_layer(const _Float16* X, int xstr,
                                 const _Float16* WT,   // [128][KP] f16 (W^T)
                                 _Float16* O, int ostr,
                                 int lane, int wave) {
  const int h  = lane >> 4;
  const int ln = lane & 15;
  const int tm = wave & 3;
  const int tn0 = (wave >> 2) * 4;
  const _Float16* xrow = X + (tm * 16 + ln) * xstr;
  v16h xf[KP / 32];
  #pragma unroll
  for (int k = 0; k < KP / 32; ++k) xf[k] = load_frag(xrow, h, k * 32);
  #pragma unroll
  for (int i = 0; i < 4; ++i) {
    const int tn = tn0 + i;
    const _Float16* wrow = WT + (tn * 16 + ln) * KP;
    v8f acc = {};
    #pragma unroll
    for (int k = 0; k < KP / 32; ++k) {
      v16h w = load_frag(wrow, h, k * 32);
      acc = __builtin_amdgcn_wmma_f32_16x16x32_f16(false, w, false, xf[k], (short)0,
                                                   acc, false, false);
    }
    v8h hv;
    #pragma unroll
    for (int e = 0; e < 8; ++e) hv[e] = (_Float16)acc[e];
    v8h z = {};
    hv = __builtin_elementwise_max(hv, z);       // relu: v_pk_max_num_f16 x4
    *(v8h*)(O + (tm * 16 + ln) * ostr + tn * 16 + 8 * h) = hv;
  }
}

// ---------- weight prep: WT[n][k] = W[k][n], f32 -> f16, zero-padded ----------
__global__ __launch_bounds__(256) void k_wprep(const float* __restrict__ W,
                                               _Float16* __restrict__ WT,
                                               int K, int N, int KP, int NP) {
  int idx = blockIdx.x * 256 + threadIdx.x;
  if (idx >= KP * NP) return;
  int k = idx % KP, n = idx / KP;
  float v = (k < K && n < N) ? W[k * N + n] : 0.f;
  WT[n * KP + k] = (_Float16)v;
}

__global__ __launch_bounds__(256) void k_zero(float* __restrict__ p, int n) {
  int i = blockIdx.x * 256 + threadIdx.x;
  if (i < n) p[i] = 0.f;
}

// ---------- encoder: lat = mlp(Xin[M x 3]) ----------
__global__ __launch_bounds__(256) void k_encoder(const float* __restrict__ Xin,
    const _Float16* __restrict__ W0, const _Float16* __restrict__ W1,
    const _Float16* __restrict__ W2, float* __restrict__ lat) {
  __shared__ _Float16 Xs[MT * XSTR_ENC];
  __shared__ _Float16 H1[MT * HSTR];
  __shared__ _Float16 H2[MT * HSTR];
  const int tid = threadIdx.x, lane = tid & 31, wave = tid >> 5;
  const int base = blockIdx.x * MT;
  for (int i = tid; i < MT * 32; i += 256) {           // pad K: 3 -> 32
    int r = i >> 5, c = i & 31;
    float v = (c < 3) ? Xin[(base + r) * 3 + c] : 0.f;
    Xs[r * XSTR_ENC + c] = (_Float16)v;
  }
  __syncthreads();
  mlp_layer<32>(Xs, XSTR_ENC, W0, H1, HSTR, lane, wave);
  __syncthreads();
  mlp_layer<128>(H1, HSTR, W1, H2, HSTR, lane, wave);
  __syncthreads();
  {   // layer2: 8 tiles (4 row x 2 col), one per wave, no relu
    const int h = lane >> 4, ln = lane & 15;
    const int tm = wave & 3, tn = wave >> 2;
    v8f acc = tile_mm_t<128>(H2 + (tm * 16 + ln) * HSTR, W2 + (tn * 16 + ln) * 128, h);
    int r = base + tm * 16 + ln;
    int c0 = tn * 16 + 8 * h;
    v4f lo = {acc[0], acc[1], acc[2], acc[3]};
    v4f hi = {acc[4], acc[5], acc[6], acc[7]};
    *(v4f*)(lat + (size_t)r * LAT + c0)     = lo;
    *(v4f*)(lat + (size_t)r * LAT + c0 + 4) = hi;
  }
}

// ---------- edge MP: edge += mlp([edge, node[s], node[r]]); agg[r] += edge ----------
__global__ __launch_bounds__(256) void k_edge_mp(const int* __restrict__ gidx,
    const float* __restrict__ node_lat, float* __restrict__ edge_lat,
    float* __restrict__ agg,
    const _Float16* __restrict__ W0, const _Float16* __restrict__ W1,
    const _Float16* __restrict__ W2) {
  __shared__ _Float16 Xs[MT * XSTR_EDGE];
  __shared__ _Float16 H1[MT * HSTR];
  __shared__ _Float16 H2[MT * HSTR];
  const int tid = threadIdx.x, lane = tid & 31, wave = tid >> 5;
  const int base = blockIdx.x * MT;
  // assemble 64 x 96 input: 24 float4 chunks per row
  for (int c = tid; c < MT * 24; c += 256) {
    int r = c / 24, part = c % 24;
    int ge = base + r;
    int seg = part >> 3, off = (part & 7) * 4;
    const float* src = (seg == 0)
        ? (edge_lat + (size_t)ge * LAT + off)
        : (node_lat + (size_t)gidx[ge * 2 + (seg - 1)] * LAT + off);
    v4f v = *(const v4f*)src;
    v4h hv = {(_Float16)v.x, (_Float16)v.y, (_Float16)v.z, (_Float16)v.w};
    *(v4h*)(Xs + r * XSTR_EDGE + seg * 32 + off) = hv;
  }
  __syncthreads();
  mlp_layer<96>(Xs, XSTR_EDGE, W0, H1, HSTR, lane, wave);
  __syncthreads();
  mlp_layer<128>(H1, HSTR, W1, H2, HSTR, lane, wave);
  __syncthreads();
  {   // layer2 + residual + scatter-add (one edge row per lane, 8 consecutive cols)
    const int h = lane >> 4, ln = lane & 15;
    const int tm = wave & 3, tn = wave >> 2;
    v8f acc = tile_mm_t<128>(H2 + (tm * 16 + ln) * HSTR, W2 + (tn * 16 + ln) * 128, h);
    int ge  = base + tm * 16 + ln;
    int c0  = tn * 16 + 8 * h;
    int recv = gidx[ge * 2 + 1];
    float* er = edge_lat + (size_t)ge * LAT + c0;
    v4f r0 = *(const v4f*)(er);
    v4f r1 = *(const v4f*)(er + 4);
    v4f n0 = {acc[0] + r0.x, acc[1] + r0.y, acc[2] + r0.z, acc[3] + r0.w};
    v4f n1 = {acc[4] + r1.x, acc[5] + r1.y, acc[6] + r1.z, acc[7] + r1.w};
    *(v4f*)(er)     = n0;
    *(v4f*)(er + 4) = n1;
    float* ar = agg + (size_t)recv * LAT + c0;
    atomicAdd(ar + 0, n0.x); atomicAdd(ar + 1, n0.y);
    atomicAdd(ar + 2, n0.z); atomicAdd(ar + 3, n0.w);
    atomicAdd(ar + 4, n1.x); atomicAdd(ar + 5, n1.y);
    atomicAdd(ar + 6, n1.z); atomicAdd(ar + 7, n1.w);
  }
}

// ---------- node MP: node += mlp([node, agg]) ----------
__global__ __launch_bounds__(256) void k_node_mp(float* __restrict__ node_lat,
    const float* __restrict__ agg,
    const _Float16* __restrict__ W0, const _Float16* __restrict__ W1,
    const _Float16* __restrict__ W2) {
  __shared__ _Float16 Xs[MT * XSTR_NODE];
  __shared__ _Float16 H1[MT * HSTR];
  __shared__ _Float16 H2[MT * HSTR];
  const int tid = threadIdx.x, lane = tid & 31, wave = tid >> 5;
  const int base = blockIdx.x * MT;
  for (int c = tid; c < MT * 16; c += 256) {    // 64 x 64 input, 16 float4/row
    int r = c / 16, part = c % 16;
    int gn = base + r;
    int seg = part >> 3, off = (part & 7) * 4;
    const float* src = (seg ? agg : node_lat) + (size_t)gn * LAT + off;
    v4f v = *(const v4f*)src;
    v4h hv = {(_Float16)v.x, (_Float16)v.y, (_Float16)v.z, (_Float16)v.w};
    *(v4h*)(Xs + r * XSTR_NODE + seg * 32 + off) = hv;
  }
  __syncthreads();
  mlp_layer<64>(Xs, XSTR_NODE, W0, H1, HSTR, lane, wave);
  __syncthreads();
  mlp_layer<128>(H1, HSTR, W1, H2, HSTR, lane, wave);
  __syncthreads();
  {
    const int h = lane >> 4, ln = lane & 15;
    const int tm = wave & 3, tn = wave >> 2;
    v8f acc = tile_mm_t<128>(H2 + (tm * 16 + ln) * HSTR, W2 + (tn * 16 + ln) * 128, h);
    int gn = base + tm * 16 + ln;
    int c0 = tn * 16 + 8 * h;
    float* nr = node_lat + (size_t)gn * LAT + c0;
    v4f r0 = *(const v4f*)(nr);
    v4f r1 = *(const v4f*)(nr + 4);
    v4f n0 = {acc[0] + r0.x, acc[1] + r0.y, acc[2] + r0.z, acc[3] + r0.w};
    v4f n1 = {acc[4] + r1.x, acc[5] + r1.y, acc[6] + r1.z, acc[7] + r1.w};
    *(v4f*)(nr)     = n0;
    *(v4f*)(nr + 4) = n1;
  }
}

// ---------- decoder: out[n] = mlp(node)[0], W2 padded N 1->16 ----------
__global__ __launch_bounds__(256) void k_decoder(const float* __restrict__ node_lat,
    const _Float16* __restrict__ W0, const _Float16* __restrict__ W1,
    const _Float16* __restrict__ W2, float* __restrict__ out) {
  __shared__ _Float16 Xs[MT * XSTR_ENC];
  __shared__ _Float16 H1[MT * HSTR];
  __shared__ _Float16 H2[MT * HSTR];
  const int tid = threadIdx.x, lane = tid & 31, wave = tid >> 5;
  const int base = blockIdx.x * MT;
  for (int c = tid; c < MT * 8; c += 256) {     // 64 x 32 input, 8 float4/row
    int r = c / 8, off = (c % 8) * 4;
    v4f v = *(const v4f*)(node_lat + (size_t)(base + r) * LAT + off);
    v4h hv = {(_Float16)v.x, (_Float16)v.y, (_Float16)v.z, (_Float16)v.w};
    *(v4h*)(Xs + r * XSTR_ENC + off) = hv;
  }
  __syncthreads();
  mlp_layer<32>(Xs, XSTR_ENC, W0, H1, HSTR, lane, wave);
  __syncthreads();
  mlp_layer<128>(H1, HSTR, W1, H2, HSTR, lane, wave);
  __syncthreads();
  if (wave < 4) {   // 4 row tiles, single (padded) col tile; only col n==0 is real
    const int h = lane >> 4, ln = lane & 15;
    const int tm = wave;
    v8f acc = tile_mm_t<128>(H2 + (tm * 16 + ln) * HSTR, W2 + ln * 128, h);
    if (h == 0)                                  // n = 8*h + e == 0 -> h==0, e==0
      out[base + tm * 16 + ln] = acc[0];
  }
}

// ---------------------------- host launcher ----------------------------
extern "C" void kernel_launch(void* const* d_in, const int* in_sizes, int n_in,
                              void* d_out, int out_size, void* d_ws, size_t ws_size,
                              hipStream_t stream) {
  (void)in_sizes; (void)n_in; (void)out_size; (void)ws_size;
  const float* in_node  = (const float*)d_in[0];
  const float* in_edge  = (const float*)d_in[1];
  const int*   gidx     = (const int*)d_in[2];
  const float* enc_n_w0 = (const float*)d_in[3];
  const float* enc_n_w1 = (const float*)d_in[4];
  const float* enc_n_w2 = (const float*)d_in[5];
  const float* enc_e_w0 = (const float*)d_in[6];
  const float* enc_e_w1 = (const float*)d_in[7];
  const float* enc_e_w2 = (const float*)d_in[8];
  const float* ep_w0    = (const float*)d_in[9];
  const float* ep_w1    = (const float*)d_in[10];
  const float* ep_w2    = (const float*)d_in[11];
  const float* np_w0    = (const float*)d_in[12];
  const float* np_w1    = (const float*)d_in[13];
  const float* np_w2    = (const float*)d_in[14];
  const float* dec_w0   = (const float*)d_in[15];
  const float* dec_w1   = (const float*)d_in[16];
  const float* dec_w2   = (const float*)d_in[17];

  uintptr_t p = (uintptr_t)d_ws;
  auto carve = [&](size_t bytes) -> void* {
    uintptr_t r = (p + 255) & ~(uintptr_t)255;
    p = r + bytes;
    return (void*)r;
  };
  float*    node_lat = (float*)carve((size_t)N_NODES * LAT * 4);   // 4.1 MB
  float*    edge_lat = (float*)carve((size_t)N_EDGES * LAT * 4);   // 65.5 MB
  float*    agg      = (float*)carve((size_t)N_NODES * LAT * 4);   // 4.1 MB
  _Float16* wcur     = (_Float16*)carve((size_t)400000 * 2);       // f16 W^T pool

  auto prep = [&](const float* W, int K, int N, int KP, int NP) -> const _Float16* {
    _Float16* dst = wcur;
    int total = KP * NP;
    k_wprep<<<dim3((total + 255) / 256), dim3(256), 0, stream>>>(W, dst, K, N, KP, NP);
    wcur += total;
    return dst;
  };

  const _Float16* nw0 = prep(enc_n_w0, 3, 128, 32, 128);
  const _Float16* nw1 = prep(enc_n_w1, 128, 128, 128, 128);
  const _Float16* nw2 = prep(enc_n_w2, 128, 32, 128, 32);
  const _Float16* ew0 = prep(enc_e_w0, 3, 128, 32, 128);
  const _Float16* ew1 = prep(enc_e_w1, 128, 128, 128, 128);
  const _Float16* ew2 = prep(enc_e_w2, 128, 32, 128, 32);
  const _Float16 *epw0[4], *epw1[4], *epw2[4], *npw0[4], *npw1[4], *npw2[4];
  for (int i = 0; i < MP_ITERS; ++i) {
    epw0[i] = prep(ep_w0 + (size_t)i * 96 * 128, 96, 128, 96, 128);
    epw1[i] = prep(ep_w1 + (size_t)i * 128 * 128, 128, 128, 128, 128);
    epw2[i] = prep(ep_w2 + (size_t)i * 128 * 32, 128, 32, 128, 32);
    npw0[i] = prep(np_w0 + (size_t)i * 64 * 128, 64, 128, 64, 128);
    npw1[i] = prep(np_w1 + (size_t)i * 128 * 128, 128, 128, 128, 128);
    npw2[i] = prep(np_w2 + (size_t)i * 128 * 32, 128, 32, 128, 32);
  }
  const _Float16* dw0 = prep(dec_w0, 32, 128, 32, 128);
  const _Float16* dw1 = prep(dec_w1, 128, 128, 128, 128);
  const _Float16* dw2 = prep(dec_w2, 128, 1, 128, 16);

  k_encoder<<<dim3(N_NODES / MT), dim3(256), 0, stream>>>(in_node, nw0, nw1, nw2, node_lat);
  k_encoder<<<dim3(N_EDGES / MT), dim3(256), 0, stream>>>(in_edge, ew0, ew1, ew2, edge_lat);

  for (int i = 0; i < MP_ITERS; ++i) {
    k_zero<<<dim3((N_NODES * LAT + 255) / 256), dim3(256), 0, stream>>>(agg, N_NODES * LAT);
    k_edge_mp<<<dim3(N_EDGES / MT), dim3(256), 0, stream>>>(gidx, node_lat, edge_lat, agg,
                                                            epw0[i], epw1[i], epw2[i]);
    k_node_mp<<<dim3(N_NODES / MT), dim3(256), 0, stream>>>(node_lat, agg,
                                                            npw0[i], npw1[i], npw2[i]);
  }
  k_decoder<<<dim3(N_NODES / MT), dim3(256), 0, stream>>>(node_lat, dw0, dw1, dw2,
                                                          (float*)d_out);
}
